// MiniCPMAttention_55929064128823
// MI455X (gfx1250) — matrix operand order repaired and checked by reading, not compiled, not run on hardware
//
#include <hip/hip_runtime.h>
#include <math.h>

typedef __attribute__((ext_vector_type(16))) __bf16 v16bf;
typedef __attribute__((ext_vector_type(8)))  float  v8f;
typedef __attribute__((ext_vector_type(4)))  unsigned int vu4;
typedef __attribute__((ext_vector_type(8)))  int  vi8;
typedef __attribute__((ext_vector_type(4)))  int  vi4;

#define HQn 32
#define HKn 8
#define DH  64
#define BB  2
#define SS  2048
#define HH  2048
#define NT  (BB*SS)              // 4096 tokens
#define NQKV ((HQn+2*HKn)*DH)    // 3072
#define KSTEP 64

#if defined(__AMDGCN__) || defined(__HIP_DEVICE_COMPILE__)
#if __has_builtin(__builtin_amdgcn_tensor_load_to_lds)
#define HAVE_TDM 1
#endif
#endif

static __device__ __forceinline__ __bf16 f2bf(float f) {
  unsigned u = __builtin_bit_cast(unsigned, f);
  unsigned r = (u + 0x7FFFu + ((u >> 16) & 1u)) >> 16;
  unsigned short s = (unsigned short)r;
  return __builtin_bit_cast(__bf16, s);
}

union BF16x16 { v16bf v; uint4 u[2]; };

#ifdef HAVE_TDM
static __device__ __forceinline__ unsigned lds_off(const void* p) {
  return (unsigned)(unsigned long long)(uintptr_t)p;   // low 32 bits of flat addr == LDS offset
}

// 2D TDM load: tile (tile_h rows x tile_w elems) of a bf16 tensor with row
// stride `stride` elems, tile origin at `gsrc`, into LDS at byte offset `lds`.
// D# layout per CDNA5 ISA.
static __device__ __forceinline__ void tdm_load_2d(
    unsigned lds, const void* gsrc, unsigned tensor_w, unsigned tensor_h,
    unsigned tile_w, unsigned tile_h, unsigned stride)
{
  unsigned long long ga = (unsigned long long)(uintptr_t)gsrc;
  vu4 g0;
  g0[0] = 1u;                                           // count=1, user mode
  g0[1] = lds;                                          // lds_addr (bytes)
  g0[2] = (unsigned)(ga & 0xFFFFFFFFu);                 // global_addr[31:0]
  g0[3] = (unsigned)((ga >> 32) & 0x01FFFFFFu) | (2u << 30); // addr[56:32], type=2
  vi8 g1;
  g1[0] = (int)(1u << 16);                              // data_size=1 -> 2 bytes
  g1[1] = (int)((tensor_w & 0xFFFFu) << 16);            // tensor_dim0[15:0]
  g1[2] = (int)(((tensor_w >> 16) & 0xFFFFu) | ((tensor_h & 0xFFFFu) << 16));
  g1[3] = (int)(((tensor_h >> 16) & 0xFFFFu) | ((tile_w & 0xFFFFu) << 16));
  g1[4] = (int)(tile_h & 0xFFFFu);                      // tile_dim1 (tile_dim2=0)
  g1[5] = (int)stride;                                  // tensor_dim0_stride[31:0]
  g1[6] = 0;                                            // stride[47:32]=0, dim1_stride lo=0
  g1[7] = 0;
  vi4 z = {0, 0, 0, 0};
#if __has_include(<hip/amd_detail/amd_gfx1250_TDM.h>)
  vi8 z8 = {0, 0, 0, 0, 0, 0, 0, 0};
  __builtin_amdgcn_tensor_load_to_lds(g0, g1, z, z, z8, 0);
#else
  __builtin_amdgcn_tensor_load_to_lds(g0, g1, z, z, 0);
#endif
}
#endif

// ---------------------------------------------------------------- convert
__global__ void cvt_f32_bf16(const float* __restrict__ in,
                             __bf16* __restrict__ out, int n) {
  int i = blockIdx.x * blockDim.x + threadIdx.x;
  if (i < n) out[i] = f2bf(in[i]);
}

// ---------------------------------------------------------------- GEMM: C[M,N] = A[M,K] * W[N,K]^T  (bf16 in, f32 out)
// Block = 8 waves -> 128x64 tile. K staged through LDS in 64-wide steps,
// double-buffered via the Tensor Data Mover (TENSORcnt), WMMAs fed by ds_load.
__global__ __launch_bounds__(256) void gemm_bf16_wmma(
    const __bf16* __restrict__ A, const __bf16* __restrict__ W,
    float* __restrict__ C, int M, int N, int K)
{
  __shared__ __bf16 As[2][128 * KSTEP];   // 16 KB x2
  __shared__ __bf16 Bs[2][64 * KSTEP];    //  8 KB x2
  const int lane = threadIdx.x & 31;
  const int wid  = threadIdx.x >> 5;
  const int lr   = lane & 15;
  const int hi   = (lane >> 4) & 1;
  const int mblk  = blockIdx.y * 128;
  const int nbase = blockIdx.x * 64;
  const int klA = hi * 16;
  const int klB = hi * 16;

#ifdef HAVE_TDM
  if (wid == 0) {
    tdm_load_2d(lds_off(&As[0][0]), A + (size_t)mblk * K, K, M, KSTEP, 128, K);
    tdm_load_2d(lds_off(&Bs[0][0]), W + (size_t)nbase * K, K, N, KSTEP, 64, K);
  }
#endif

  v8f acc0 = {}, acc1 = {}, acc2 = {}, acc3 = {};

  const int nsteps = K / KSTEP;
  for (int s = 0; s < nsteps; ++s) {
    const int p = s & 1;
#ifdef HAVE_TDM
    if (wid == 0) {
      if (s + 1 < nsteps) {
        const int kn = (s + 1) * KSTEP;
        tdm_load_2d(lds_off(&As[1 - p][0]), A + (size_t)mblk * K + kn, K, M, KSTEP, 128, K);
        tdm_load_2d(lds_off(&Bs[1 - p][0]), W + (size_t)nbase * K + kn, K, N, KSTEP, 64, K);
        __builtin_amdgcn_s_wait_tensorcnt(2);   // buffer p complete, next 2 in flight
      } else {
        __builtin_amdgcn_s_wait_tensorcnt(0);
      }
    }
    __syncthreads();                            // buffer p visible to all waves
#else
    // Fallback: cooperative global->LDS copy of buffer p.
    {
      const int k0 = s * KSTEP;
      for (int idx = threadIdx.x; idx < 128 * KSTEP / 8; idx += 256) {
        int row = idx / (KSTEP / 8), col = (idx % (KSTEP / 8)) * 8;
        *(uint4*)&As[p][row * KSTEP + col] =
            *(const uint4*)(A + (size_t)(mblk + row) * K + k0 + col);
      }
      for (int idx = threadIdx.x; idx < 64 * KSTEP / 8; idx += 256) {
        int row = idx / (KSTEP / 8), col = (idx % (KSTEP / 8)) * 8;
        *(uint4*)&Bs[p][row * KSTEP + col] =
            *(const uint4*)(W + (size_t)(nbase + row) * K + k0 + col);
      }
      __syncthreads();
    }
#endif

    const __bf16* a  = &As[p][0] + (size_t)(wid * 16 + lr) * KSTEP;
    const __bf16* b0 = &Bs[p][0] + (size_t)(0  + lr) * KSTEP;
    const __bf16* b1 = &Bs[p][0] + (size_t)(16 + lr) * KSTEP;
    const __bf16* b2 = &Bs[p][0] + (size_t)(32 + lr) * KSTEP;
    const __bf16* b3 = &Bs[p][0] + (size_t)(48 + lr) * KSTEP;

#pragma unroll
    for (int d = 0; d < KSTEP; d += 32) {
      BF16x16 af;
      af.u[0] = *(const uint4*)(a + d + klA);
      af.u[1] = *(const uint4*)(a + d + klA + 8);
      BF16x16 bf0, bf1, bf2, bf3;
      bf0.u[0] = *(const uint4*)(b0 + d + klB);
      bf0.u[1] = *(const uint4*)(b0 + d + klB + 8);
      bf1.u[0] = *(const uint4*)(b1 + d + klB);
      bf1.u[1] = *(const uint4*)(b1 + d + klB + 8);
      bf2.u[0] = *(const uint4*)(b2 + d + klB);
      bf2.u[1] = *(const uint4*)(b2 + d + klB + 8);
      bf3.u[0] = *(const uint4*)(b3 + d + klB);
      bf3.u[1] = *(const uint4*)(b3 + d + klB + 8);
      acc0 = __builtin_amdgcn_wmma_f32_16x16x32_bf16(false, af.v, false, bf0.v, (short)0, acc0, false, false);
      acc1 = __builtin_amdgcn_wmma_f32_16x16x32_bf16(false, af.v, false, bf1.v, (short)0, acc1, false, false);
      acc2 = __builtin_amdgcn_wmma_f32_16x16x32_bf16(false, af.v, false, bf2.v, (short)0, acc2, false, false);
      acc3 = __builtin_amdgcn_wmma_f32_16x16x32_bf16(false, af.v, false, bf3.v, (short)0, acc3, false, false);
    }
    __syncthreads();                            // all waves done reading buffer p
  }

#pragma unroll
  for (int i = 0; i < 8; ++i) {
    int row = mblk + wid * 16 + i + hi * 8;
    float* crow = C + (size_t)row * N + nbase + lr;
    crow[0]  = acc0[i];
    crow[16] = acc1[i];
    crow[32] = acc2[i];
    crow[48] = acc3[i];
  }
}

// ---------------------------------------------------------------- RMSNorm + NeoX RoPE + V transpose
__global__ __launch_bounds__(256) void normrope(
    const float* __restrict__ qkv, const int* __restrict__ pos,
    const float* __restrict__ qw, const float* __restrict__ kw,
    __bf16* __restrict__ Q, __bf16* __restrict__ Kc, __bf16* __restrict__ Vt)
{
  const int lane = threadIdx.x & 31;
  const int wid  = threadIdx.x >> 5;
  int job = blockIdx.x * 8 + wid;
  int t = job / 48;
  int h = job % 48;
  int b = t / SS, s = t % SS;

  int off;
  if (h < HQn)            off = h * DH;
  else if (h < HQn + HKn) off = HQn * DH + (h - HQn) * DH;
  else                    off = HQn * DH + HKn * DH + (h - HQn - HKn) * DH;

  const float* row = qkv + (size_t)t * NQKV + off;
  float x1 = row[lane];
  float x2 = row[lane + 32];

  if (h < HQn + HKn) {
    float ss2 = x1 * x1 + x2 * x2;
    ss2 += __shfl_xor(ss2, 1, 32);
    ss2 += __shfl_xor(ss2, 2, 32);
    ss2 += __shfl_xor(ss2, 4, 32);
    ss2 += __shfl_xor(ss2, 8, 32);
    ss2 += __shfl_xor(ss2, 16, 32);
    float r = rsqrtf(ss2 * (1.0f / 64.0f) + 1e-5f);
    const float* nw = (h < HQn) ? qw : kw;
    float y1 = x1 * r * nw[lane];
    float y2 = x2 * r * nw[lane + 32];
    float p = (float)pos[t];
    float invf = __powf(10000.0f, -(float)lane * (1.0f / 32.0f));
    float ang = p * invf;
    float c = __cosf(ang), sn = __sinf(ang);
    float o1 = y1 * c - y2 * sn;
    float o2 = y2 * c + y1 * sn;
    if (h < HQn) {
      __bf16* d = Q + ((size_t)(b * HQn + h) * SS + s) * DH;
      d[lane] = f2bf(o1); d[lane + 32] = f2bf(o2);
    } else {
      __bf16* d = Kc + ((size_t)(b * HKn + (h - HQn)) * SS + s) * DH;
      d[lane] = f2bf(o1); d[lane + 32] = f2bf(o2);
    }
  } else {
    int hv = h - HQn - HKn;
    __bf16* d = Vt + (size_t)(b * HKn + hv) * DH * SS;   // [D, S]
    d[(size_t)lane * SS + s]        = f2bf(x1);
    d[(size_t)(lane + 32) * SS + s] = f2bf(x2);
  }
}

// ---------------------------------------------------------------- causal GQA flash attention
__global__ __launch_bounds__(128) void flash_attn(
    const __bf16* __restrict__ Q, const __bf16* __restrict__ Kc,
    const __bf16* __restrict__ Vt, __bf16* __restrict__ O)
{
  __shared__ __bf16 lp[4][16 * 32];
  const int lane = threadIdx.x & 31;
  const int wid  = threadIdx.x >> 5;
  const int lr   = lane & 15;
  const int hi   = (lane >> 4) & 1;

  int job = blockIdx.x * 4 + wid;
  int qt = job & 127;
  int hq = (job >> 7) & 31;
  int b  = job >> 12;
  int hk = hq >> 2;
  const int qbase = qt * 16;

  const __bf16* Qh = Q  + (size_t)(b * HQn + hq) * SS * DH;
  const __bf16* Kh = Kc + (size_t)(b * HKn + hk) * SS * DH;
  const __bf16* Vh = Vt + (size_t)(b * HKn + hk) * DH * SS;

  BF16x16 aq0, aq1;
  {
    const __bf16* qr = Qh + (size_t)(qbase + lr) * DH;
    int kl = hi * 16;
    aq0.u[0] = *(const uint4*)(qr + 0  + kl);
    aq0.u[1] = *(const uint4*)(qr + 8 + kl);
    aq1.u[0] = *(const uint4*)(qr + 32 + kl);
    aq1.u[1] = *(const uint4*)(qr + 40 + kl);
  }

  float m[8], l[8];
  v8f o0 = {}, o1a = {}, o2a = {}, o3a = {};
#pragma unroll
  for (int i = 0; i < 8; ++i) { m[i] = -1e30f; l[i] = 0.f; }

  const float scale = 0.125f;
  const int klB = hi * 16;

  for (int kb = 0; kb <= qbase + 15; kb += 32) {
    // ---- scores
    v8f s0 = {}, s1 = {};
    {
      BF16x16 bk;
      const __bf16* kr0 = Kh + (size_t)(kb + lr) * DH;
      bk.u[0] = *(const uint4*)(kr0 + klB);
      bk.u[1] = *(const uint4*)(kr0 + klB + 8);
      s0 = __builtin_amdgcn_wmma_f32_16x16x32_bf16(false, aq0.v, false, bk.v, (short)0, s0, false, false);
      bk.u[0] = *(const uint4*)(kr0 + 32 + klB);
      bk.u[1] = *(const uint4*)(kr0 + 32 + klB + 8);
      s0 = __builtin_amdgcn_wmma_f32_16x16x32_bf16(false, aq1.v, false, bk.v, (short)0, s0, false, false);
      const __bf16* kr1 = Kh + (size_t)(kb + 16 + lr) * DH;
      bk.u[0] = *(const uint4*)(kr1 + klB);
      bk.u[1] = *(const uint4*)(kr1 + klB + 8);
      s1 = __builtin_amdgcn_wmma_f32_16x16x32_bf16(false, aq0.v, false, bk.v, (short)0, s1, false, false);
      bk.u[0] = *(const uint4*)(kr1 + 32 + klB);
      bk.u[1] = *(const uint4*)(kr1 + 32 + klB + 8);
      s1 = __builtin_amdgcn_wmma_f32_16x16x32_bf16(false, aq1.v, false, bk.v, (short)0, s1, false, false);
    }

    // ---- prefetch V fragments early: their latency overlaps the softmax VALU
    BF16x16 bv0, bv1, bv2, bv3;
    {
      const __bf16* vr = Vh + (size_t)lr * SS + kb + klB;
      bv0.u[0] = *(const uint4*)(vr);
      bv0.u[1] = *(const uint4*)(vr + 8);
      vr = Vh + (size_t)(16 + lr) * SS + kb + klB;
      bv1.u[0] = *(const uint4*)(vr);
      bv1.u[1] = *(const uint4*)(vr + 8);
      vr = Vh + (size_t)(32 + lr) * SS + kb + klB;
      bv2.u[0] = *(const uint4*)(vr);
      bv2.u[1] = *(const uint4*)(vr + 8);
      vr = Vh + (size_t)(48 + lr) * SS + kb + klB;
      bv3.u[0] = *(const uint4*)(vr);
      bv3.u[1] = *(const uint4*)(vr + 8);
    }

    // ---- scale + causal mask
    const bool needMask = (kb + 31 > qbase);
    float p0[8], p1[8], sc[8];
#pragma unroll
    for (int i = 0; i < 8; ++i) {
      float v0 = s0[i] * scale;
      float v1 = s1[i] * scale;
      if (needMask) {
        int qr = qbase + i + hi * 8;
        if (kb + lr > qr)      v0 = -1e30f;
        if (kb + 16 + lr > qr) v1 = -1e30f;
      }
      p0[i] = v0; p1[i] = v1;
    }

    // ---- online softmax
#pragma unroll
    for (int i = 0; i < 8; ++i) {
      float mx = fmaxf(p0[i], p1[i]);
      mx = fmaxf(mx, __shfl_xor(mx, 1, 32));
      mx = fmaxf(mx, __shfl_xor(mx, 2, 32));
      mx = fmaxf(mx, __shfl_xor(mx, 4, 32));
      mx = fmaxf(mx, __shfl_xor(mx, 8, 32));
      float mn = fmaxf(m[i], mx);
      sc[i] = __expf(m[i] - mn);
      m[i] = mn;
      p0[i] = __expf(p0[i] - mn);
      p1[i] = __expf(p1[i] - mn);
      float rs = p0[i] + p1[i];
      rs += __shfl_xor(rs, 1, 32);
      rs += __shfl_xor(rs, 2, 32);
      rs += __shfl_xor(rs, 4, 32);
      rs += __shfl_xor(rs, 8, 32);
      l[i] = l[i] * sc[i] + rs;
      o0[i] *= sc[i]; o1a[i] *= sc[i]; o2a[i] *= sc[i]; o3a[i] *= sc[i];
    }

    // ---- P (C-layout) -> LDS -> A-layout fragment
    __bf16* sp = lp[wid];
#pragma unroll
    for (int i = 0; i < 8; ++i) {
      int row = i + hi * 8;
      sp[row * 32 + lr]      = f2bf(p0[i]);
      sp[row * 32 + 16 + lr] = f2bf(p1[i]);
    }
    BF16x16 ap;
    {
      int kl = hi * 16;
      ap.u[0] = *(const uint4*)(sp + lr * 32 + kl);
      ap.u[1] = *(const uint4*)(sp + lr * 32 + 8 + kl);
    }

    // ---- O += P x V
    o0  = __builtin_amdgcn_wmma_f32_16x16x32_bf16(false, ap.v, false, bv0.v, (short)0, o0,  false, false);
    o1a = __builtin_amdgcn_wmma_f32_16x16x32_bf16(false, ap.v, false, bv1.v, (short)0, o1a, false, false);
    o2a = __builtin_amdgcn_wmma_f32_16x16x32_bf16(false, ap.v, false, bv2.v, (short)0, o2a, false, false);
    o3a = __builtin_amdgcn_wmma_f32_16x16x32_bf16(false, ap.v, false, bv3.v, (short)0, o3a, false, false);
  }

#pragma unroll
  for (int i = 0; i < 8; ++i) {
    int qr = qbase + i + hi * 8;
    float inv = 1.0f / l[i];
    __bf16* orow = O + (size_t)(b * SS + qr) * (HQn * DH) + hq * DH;
    orow[0  + lr] = f2bf(o0[i]  * inv);
    orow[16 + lr] = f2bf(o1a[i] * inv);
    orow[32 + lr] = f2bf(o2a[i] * inv);
    orow[48 + lr] = f2bf(o3a[i] * inv);
  }
}

// ----------------------------------------------------------------
extern "C" void kernel_launch(void* const* d_in, const int* in_sizes, int n_in,
                              void* d_out, int out_size, void* d_ws, size_t ws_size,
                              hipStream_t stream) {
  const int*   positions = (const int*)d_in[0];
  const float* hidden    = (const float*)d_in[1];
  const float* w_qkv     = (const float*)d_in[2];
  const float* w_o       = (const float*)d_in[3];
  const float* q_norm_w  = (const float*)d_in[4];
  const float* k_norm_w  = (const float*)d_in[5];

  char* p = (char*)d_ws;
  __bf16* Xb  = (__bf16*)p;  p += (size_t)NT * HH * 2;
  __bf16* Wqb = (__bf16*)p;  p += (size_t)NQKV * HH * 2;
  __bf16* Wob = (__bf16*)p;  p += (size_t)HH * HH * 2;
  float*  QKV = (float*)p;   p += (size_t)NT * NQKV * 4;
  __bf16* Qb  = (__bf16*)p;  p += (size_t)NT * HQn * DH * 2;
  __bf16* Kb  = (__bf16*)p;  p += (size_t)NT * HKn * DH * 2;
  __bf16* Vtb = (__bf16*)p;  p += (size_t)NT * HKn * DH * 2;
  __bf16* Attn = Xb;   // alias: X dead after the QKV GEMM
  float* out = (float*)d_out;

  int n1 = NT * HH;
  cvt_f32_bf16<<<(n1 + 255) / 256, 256, 0, stream>>>(hidden, Xb, n1);
  int n2 = NQKV * HH;
  cvt_f32_bf16<<<(n2 + 255) / 256, 256, 0, stream>>>(w_qkv, Wqb, n2);
  int n3 = HH * HH;
  cvt_f32_bf16<<<(n3 + 255) / 256, 256, 0, stream>>>(w_o, Wob, n3);

  gemm_bf16_wmma<<<dim3(NQKV / 64, NT / 128), 256, 0, stream>>>(Xb, Wqb, QKV, NT, NQKV, HH);

  normrope<<<(NT * 48) / 8, 256, 0, stream>>>(QKV, positions, q_norm_w, k_norm_w, Qb, Kb, Vtb);

  flash_attn<<<(BB * HQn * (SS / 16)) / 4, 128, 0, stream>>>(Qb, Kb, Vtb, Attn);

  gemm_bf16_wmma<<<dim3(HH / 64, NT / 128), 256, 0, stream>>>(Attn, Wob, out, NT, HH, HH);
}
